// RWKV_TimeMix_62921270886835
// MI455X (gfx1250) — matrix operand order repaired and checked by reading, not compiled, not run on hardware
//
#include <hip/hip_runtime.h>

typedef __attribute__((ext_vector_type(16))) _Float16 v16h;
typedef __attribute__((ext_vector_type(8)))  _Float16 v8h;
typedef __attribute__((ext_vector_type(8)))  float    v8f;

#define BB 32
#define TT 1024
#define CC 768
#define MM (BB*TT)   // 32768 rows

// ---------------- f32 -> f16 convert (weights) ----------------
__global__ void f32_to_f16_kernel(const float* __restrict__ s, _Float16* __restrict__ d, int n) {
    int i = blockIdx.x * blockDim.x + threadIdx.x;
    if (i < n) d[i] = (_Float16)s[i];
}

// ---------------- time-shift mix: out = xx + tm*(x - xx), xx = x[t-1] (0 at t==0) ----
__global__ void mix_kernel(const float* __restrict__ x, const float* __restrict__ tm,
                           _Float16* __restrict__ out) {
    int i = blockIdx.x * blockDim.x + threadIdx.x;   // over MM*CC (< 2^31)
    int c   = i % CC;
    int row = i / CC;
    int t   = row % TT;
    float xv = x[i];
    float xx = (t == 0) ? 0.0f : x[i - CC];
    float m  = tm[c];
    out[i] = (_Float16)(xx + m * (xv - xx));
}

// ---------------- WMMA GEMM: C[m,n] = sum_k A[m,k] * W[n,k] ----------------
// A: f16 row-major M x K ; W: f16 row-major N x K ; output f32 or f16 (row-major M x N)
// Block = 256 threads = 8 waves, block tile 256(M) x 64(N); wave tile 32 x 64.
template <typename OutT>
__global__ __launch_bounds__(256) void gemm_f16_wmma(const _Float16* __restrict__ A,
                                                     const _Float16* __restrict__ W,
                                                     OutT* __restrict__ Cg,
                                                     int M, int N, int K) {
    const int lane = threadIdx.x & 31;
    const int wave = threadIdx.x >> 5;
    const int l15  = lane & 15;
    const int hi   = lane >> 4;           // 0 for lanes 0-15, 1 for lanes 16-31
    const int m0   = blockIdx.x * 256 + wave * 32;
    const int n0   = blockIdx.y * 64;

    // A fragment: lane holds row (m0+mf*16+l15); halves K = base..base+15
    const _Float16* aBase[2];
#pragma unroll
    for (int mf = 0; mf < 2; ++mf)
        aBase[mf] = A + (size_t)(m0 + mf * 16 + l15) * K + hi * 16;

    // B fragment: lane holds W-row (= output col) n0+nf*16+l15; halves K = base..base+15
    const _Float16* bBase[4];
#pragma unroll
    for (int nf = 0; nf < 4; ++nf)
        bBase[nf] = W + (size_t)(n0 + nf * 16 + l15) * K + hi * 16;

    v8f acc[2][4];
#pragma unroll
    for (int mf = 0; mf < 2; ++mf)
#pragma unroll
        for (int nf = 0; nf < 4; ++nf)
            acc[mf][nf] = {};

    for (int k0 = 0; k0 < K; k0 += 32) {
        v16h a[2];
#pragma unroll
        for (int mf = 0; mf < 2; ++mf) {
            v8h lo = *(const v8h*)(aBase[mf] + k0);
            v8h hh = *(const v8h*)(aBase[mf] + k0 + 8);
            a[mf] = __builtin_shufflevector(lo, hh, 0, 1, 2, 3, 4, 5, 6, 7,
                                            8, 9, 10, 11, 12, 13, 14, 15);
        }
        v16h b[4];
#pragma unroll
        for (int nf = 0; nf < 4; ++nf)
            b[nf] = *(const v16h*)(bBase[nf] + k0);

#pragma unroll
        for (int mf = 0; mf < 2; ++mf)
#pragma unroll
            for (int nf = 0; nf < 4; ++nf)
                acc[mf][nf] = __builtin_amdgcn_wmma_f32_16x16x32_f16(
                    /*neg_a=*/false, a[mf], /*neg_b=*/false, b[nf],
                    /*c_mod=*/(short)0, acc[mf][nf],
                    /*reuse_a=*/false, /*reuse_b=*/false);
    }

    // Store: VGPR j holds row m0+mf*16+8*hi+j, col n0+nf*16+l15
#pragma unroll
    for (int mf = 0; mf < 2; ++mf)
#pragma unroll
        for (int nf = 0; nf < 4; ++nf) {
            const int col = n0 + nf * 16 + l15;
#pragma unroll
            for (int j = 0; j < 8; ++j) {
                const int row = m0 + mf * 16 + 8 * hi + j;
                Cg[(size_t)row * N + col] = (OutT)acc[mf][nf][j];
            }
        }
}

// ---------------- WKV scan (numerically stable), fused with sigmoid(r)*y -> f16 ----
__global__ void wkv_sig_kernel(const _Float16* __restrict__ kf, const _Float16* __restrict__ vf,
                               const _Float16* __restrict__ rf, const float* __restrict__ td,
                               const float* __restrict__ tfirst, _Float16* __restrict__ outh) {
    int id = blockIdx.x * blockDim.x + threadIdx.x;  // over BB*CC channels
    int c = id % CC;
    int b = id / CC;
    const float w = td[c];
    const float u = tfirst[c];
    float p = 0.0f, q = 0.0f, o = -1e38f;
    size_t base = (size_t)b * TT * CC + c;
    for (int t = 0; t < TT; ++t) {
        size_t idx = base + (size_t)t * CC;
        float kt = (float)kf[idx];
        float vt = (float)vf[idx];
        float no = fmaxf(o, u + kt);
        float Aa = __expf(o - no);
        float Bc = __expf(u + kt - no);
        float y  = (Aa * p + Bc * vt) / (Aa * q + Bc);
        float no2 = fmaxf(w + o, kt);
        float A2 = __expf(w + o - no2);
        float B2 = __expf(kt - no2);
        p = A2 * p + B2 * vt;
        q = A2 * q + B2;
        o = no2;
        float r  = (float)rf[idx];
        float sr = 1.0f / (1.0f + __expf(-r));
        outh[idx] = (_Float16)(sr * y);
    }
}

extern "C" void kernel_launch(void* const* d_in, const int* in_sizes, int n_in,
                              void* d_out, int out_size, void* d_ws, size_t ws_size,
                              hipStream_t stream) {
    const float* x      = (const float*)d_in[0];
    const float* td     = (const float*)d_in[1];
    const float* tfirst = (const float*)d_in[2];
    const float* tmix[3] = { (const float*)d_in[3], (const float*)d_in[4], (const float*)d_in[5] };
    const float* Wf[4]   = { (const float*)d_in[6], (const float*)d_in[7],
                             (const float*)d_in[8], (const float*)d_in[9] };
    float* out = (float*)d_out;

    const size_t CCn = (size_t)CC * CC;       // 589,824
    const size_t MCn = (size_t)MM * CC;       // 25,165,824

    // Workspace layout (~206 MB total):
    //  [0]            4*CCn f16 : Wk,Wv,Wr,Wo in f16                       (4.7 MB)
    //  [+4*CCn]       MCn   f16 : mix buffer, later reused for sigmoid(r)*y (50 MB)
    //  [then]         3*MCn f16 : k, v, r                                  (151 MB)
    _Float16* Wh = (_Float16*)d_ws;
    _Float16* xh = Wh + 4 * CCn;
    _Float16* kf = xh + MCn;
    _Float16* vf = kf + MCn;
    _Float16* rf = vf + MCn;
    _Float16* kvr[3] = { kf, vf, rf };

    // 1) convert weights to f16
    for (int j = 0; j < 4; ++j)
        f32_to_f16_kernel<<<(int)((CCn + 255) / 256), 256, 0, stream>>>(Wf[j], Wh + j * CCn, (int)CCn);

    dim3 ggrid(MM / 256, CC / 64);   // (128, 12)

    // 2) mix -> f16, then WMMA GEMM -> k / v / r (f16 out)
    for (int j = 0; j < 3; ++j) {
        mix_kernel<<<(int)(MCn / 256), 256, 0, stream>>>(x, tmix[j], xh);
        gemm_f16_wmma<_Float16><<<ggrid, 256, 0, stream>>>(xh, Wh + j * CCn, kvr[j], MM, CC, CC);
    }

    // 3) WKV scan fused with sigmoid(r)*y -> f16 into xh
    wkv_sig_kernel<<<(BB * CC) / 256, 256, 0, stream>>>(kf, vf, rf, td, tfirst, xh);

    // 4) output GEMM (f32 out to d_out): out = (sr*y) @ Wo^T
    gemm_f16_wmma<float><<<ggrid, 256, 0, stream>>>(xh, Wh + 3 * CCn, out, MM, CC, CC);
}
